// GridTorch_1657857376697
// MI455X (gfx1250) — matrix operand order repaired and checked by reading, not compiled, not run on hardware
//
#include <hip/hip_runtime.h>
#include <hip/hip_bf16.h>

typedef __attribute__((ext_vector_type(16))) __bf16 v16bf;
typedef __attribute__((ext_vector_type(8)))  __bf16 v8bf;
typedef __attribute__((ext_vector_type(8)))  float  v8f;
typedef __attribute__((ext_vector_type(4)))  unsigned int u32x4;
typedef __attribute__((ext_vector_type(8)))  int i32x8;
typedef __attribute__((ext_vector_type(4)))  int i32x4;

#define T_STEPS 100
#define BATCH   2048
#define HS      128

// ---- packed-weight workspace layout (bf16 element offsets) ----
// B-fragment-packed tiles: tile index = kt*NT + nt ; within tile:
//   packed[(tile*32 + lane)*16 + e] = W[kt*32 + (lane<16?0:16) + e][nt*16 + (lane&15)]
#define OFF_U    0L        // 4x32 tiles  (K=128,N=512)  65536
#define OFF_WB   65536L    // 4x16 tiles  (K=128,N=256)  32768
#define OFF_WPC  98304L    // 8x16 tiles  (K=256,N=256)  65536
#define OFF_WHD  163840L   // 8x1  tiles  (K=256,N=12->16) 4096
#define OFF_WH   167936L   // 9x8  tiles  (K=268->288,N=128) 36864
#define OFF_WC   204800L   // 9x8  tiles                     36864
#define OFF_INIT 241664L   // init bf16 row-major [2048][288] 589824
#define WS_BF16_TOTAL 831488L

#define LDS_DYN_BYTES 262144  // 128KB U + 128KB Wpc staged in LDS via TDM

union BFrag { v16bf v; v8bf h[2]; };

__device__ __forceinline__ v8f wmma_bf16(v16bf a, v16bf b, v8f c) {
  return __builtin_amdgcn_wmma_f32_16x16x32_bf16(false, a, false, b, (short)0, c, false, false);
}

// A operand, contiguous k order (as packed B): lane<16 -> row=lane, k = k0+[0..15]
//                                 lane>=16 -> row=lane-16, k = k0+16+[0..15]
__device__ __forceinline__ v16bf load_a_frag(const __bf16* base, int ld, int lane, int k0) {
  int r = lane & 15;
  int koff = (lane < 16) ? 0 : 16;
  const __bf16* p = base + r * ld + k0 + koff;
  BFrag f;
  f.h[0] = *(const v8bf*)(p);
  f.h[1] = *(const v8bf*)(p + 8);
  return f.v;
}

// B operand from fragment-packed buffer (global or LDS): 32 bytes contiguous per lane
__device__ __forceinline__ v16bf load_b_frag(const __bf16* packed, int tile, int lane) {
  const __bf16* p = packed + ((long)tile * 32 + lane) * 16;
  BFrag f;
  f.h[0] = *(const v8bf*)(p);
  f.h[1] = *(const v8bf*)(p + 8);
  return f.v;
}

__device__ __forceinline__ float sigmf(float x) { return 1.0f / (1.0f + __expf(-x)); }
__device__ __forceinline__ float tanhfast(float x) {
  float e = __expf(-2.0f * x);
  return (1.0f - e) / (1.0f + e);
}

// ---- TDM: 1D bulk copy global->LDS as a 2D D# with one row of n_qwords 8-byte elems ----
__device__ __forceinline__ void tdm_load_1d(unsigned long long gaddr, unsigned lds_off,
                                            unsigned n_qwords) {
  u32x4 g0;
  g0[0] = 1u;                                         // count=1, no gather, is_restore=0
  g0[1] = lds_off;                                    // lds_addr [63:32]
  g0[2] = (unsigned)(gaddr & 0xFFFFFFFFu);            // global_addr [95:64]
  g0[3] = (unsigned)((gaddr >> 32) & 0x01FFFFFFu)     // global_addr [120:96]
        | (2u << 30);                                 // type=2 ("image") [127:126]
  i32x8 g1;
  g1[0] = (3 << 16);                                  // data_size=3 (8B); mask=0; no pad
  g1[1] = (int)((n_qwords & 0xFFFFu) << 16);          // tensor_dim0[15:0]  @ bits 63:48
  g1[2] = (int)(((n_qwords >> 16) & 0xFFFFu)          // tensor_dim0[31:16] @ bits 79:64
        | (1u << 16));                                // tensor_dim1=1      @ bits 111:80
  g1[3] = (int)(n_qwords << 16);                      // tile_dim0          @ bits 127:112
  g1[4] = 1;                                          // tile_dim1=1; tile_dim2=0
  g1[5] = (int)n_qwords;                              // tensor_dim0_stride[31:0]
  g1[6] = 0;                                          // stride hi / dim1_stride lo
  g1[7] = 0;
  i32x4 g2 = {0, 0, 0, 0};
  i32x4 g3 = {0, 0, 0, 0};
  i32x8 g4 = {0, 0, 0, 0, 0, 0, 0, 0};                // 6-arg toolchain form
  __builtin_amdgcn_tensor_load_to_lds(g0, g1, g2, g3, g4, 0);
}

// ---------------- one-time weight packing (fp32 -> bf16, fragment order) ----------------
__device__ __forceinline__ void pack_one(const float* src, int K, int Nact, int NT,
                                         __bf16* dst, long idx) {
  int e    = (int)(idx & 15);
  int lane = (int)((idx >> 4) & 31);
  long tile = idx >> 9;
  int nt = (int)(tile % NT);
  int kt = (int)(tile / NT);
  int k = kt * 32 + ((lane < 16) ? 0 : 16) + e;
  int n = nt * 16 + (lane & 15);
  float v = (k < K && n < Nact) ? src[(long)k * Nact + n] : 0.0f;
  dst[idx] = (__bf16)v;
}

__global__ void pack_weights(const float* U, const float* Wb, const float* Wpc, const float* Whd,
                             const float* Wh, const float* Wc, const float* init, __bf16* ws) {
  long idx = (long)blockIdx.x * blockDim.x + threadIdx.x;
  if (idx < 65536) { pack_one(U,   128, 512, 32, ws + OFF_U,   idx); return; }
  idx -= 65536;
  if (idx < 32768) { pack_one(Wb,  128, 256, 16, ws + OFF_WB,  idx); return; }
  idx -= 32768;
  if (idx < 65536) { pack_one(Wpc, 256, 256, 16, ws + OFF_WPC, idx); return; }
  idx -= 65536;
  if (idx < 4096)  { pack_one(Whd, 256,  12,  1, ws + OFF_WHD, idx); return; }
  idx -= 4096;
  if (idx < 36864) { pack_one(Wh,  268, 128,  8, ws + OFF_WH,  idx); return; }
  idx -= 36864;
  if (idx < 36864) { pack_one(Wc,  268, 128,  8, ws + OFF_WC,  idx); return; }
  idx -= 36864;
  if (idx < 589824) {                       // init: [2048][268] -> bf16 [2048][288] (K-pad)
    long row = idx / 288;
    int  k   = (int)(idx % 288);
    float v = (k < 268) ? init[row * 268 + k] : 0.0f;
    ws[OFF_INIT + idx] = (__bf16)v;
  }
}

// ---------------- persistent LSTM kernel: 1 block = 16 batch rows, all 100 steps ----------------
__global__ __launch_bounds__(256) void lstm_main(
    const float* __restrict__ x,    // [T,B,3]
    const float* __restrict__ W,    // [3,512]
    const float* __restrict__ b,    // [512]
    const float* __restrict__ bh,   // [128]
    const float* __restrict__ bc,   // [128]
    const float* __restrict__ bpc,  // [256]
    const float* __restrict__ bhd,  // [12]
    const __bf16* __restrict__ ws,
    float* __restrict__ out) {
  __shared__ __attribute__((aligned(16))) __bf16 hBF[16][HS];    // 4KB  h_t (bf16)
  __shared__ __attribute__((aligned(16))) __bf16 boBF[16][256];  // 8KB  bottleneck (bf16)
  __shared__ __attribute__((aligned(16))) float xs[16][4];       // 256B x_t staging
  extern __shared__ __align__(16) char dynLDS[];                 // 256KB: U + Wpc (TDM-staged)

  const int w     = threadIdx.x >> 5;   // wave 0..7 : owns hidden cols [16w,16w+16)
  const int lane  = threadIdx.x & 31;
  const int m0    = blockIdx.x * 16;    // batch row base
  const int col16 = lane & 15;
  const bool rlo  = (lane < 16);

  const __bf16* Wb_p  = ws + OFF_WB;
  const __bf16* Whd_p = ws + OFF_WHD;
  const __bf16* Wh_p  = ws + OFF_WH;
  const __bf16* Wc_p  = ws + OFF_WC;
  const __bf16* initB = ws + OFF_INIT + (long)m0 * 288;

  const __bf16* U_lds   = (const __bf16*)dynLDS;             // 65536 bf16
  const __bf16* Wpc_lds = (const __bf16*)(dynLDS + 131072);  // 65536 bf16

  const long off_pc = (long)T_STEPS * BATCH * 12;
  const long off_bo = off_pc + (long)T_STEPS * BATCH * 256;
  const long off_h  = off_bo + (long)T_STEPS * BATCH * 256;
  const long off_c  = off_h  + (long)T_STEPS * BATCH * 128;

  // ---- kick off TDM staging of U and Wpc into LDS (overlaps with h0/c0 WMMAs) ----
  if (w == 0) {
    unsigned dynOff = (__builtin_amdgcn_groupstaticsize() + 15u) & ~15u;  // dyn LDS base
    tdm_load_1d((unsigned long long)(const void*)(ws + OFF_U),   dynOff,           16384u);
    tdm_load_1d((unsigned long long)(const void*)(ws + OFF_WPC), dynOff + 131072u, 16384u);
  }

  // ---- prologue: h0 = init@Wh + bh, c0 = init@Wc + bc (wave w -> hidden ntile w) ----
  v8f creg, hreg;
  {
    float bhv = bh[w * 16 + col16];
    float bcv = bc[w * 16 + col16];
    v8f hacc, cacc;
#pragma unroll
    for (int v = 0; v < 8; v++) { hacc[v] = bhv; cacc[v] = bcv; }
#pragma unroll
    for (int kt = 0; kt < 9; kt++) {
      v16bf a = load_a_frag(initB, 288, lane, kt * 32);
      hacc = wmma_bf16(a, load_b_frag(Wh_p, kt * 8 + w, lane), hacc);
      cacc = wmma_bf16(a, load_b_frag(Wc_p, kt * 8 + w, lane), cacc);
    }
    creg = cacc;
    hreg = hacc;
  }
#pragma unroll
  for (int v = 0; v < 8; v++) {
    int r = rlo ? v : v + 8;
    hBF[r][w * 16 + col16] = (__bf16)hreg[v];
  }

  // per-lane constants for the time loop
  float bg[4], Wg0[4], Wg1[4], Wg2[4];
#pragma unroll
  for (int X = 0; X < 4; X++) {              // gate group X, ntile = X*8+w
    int col = X * 128 + w * 16 + col16;
    bg[X] = b[col];
    Wg0[X] = W[0 * 512 + col];
    Wg1[X] = W[1 * 512 + col];
    Wg2[X] = W[2 * 512 + col];
  }
  float bpcv[2];
  bpcv[0] = bpc[(2 * w + 0) * 16 + col16];
  bpcv[1] = bpc[(2 * w + 1) * 16 + col16];
  float bhdv = (w == 0 && col16 < 12) ? bhd[col16] : 0.0f;

  // TDM must have landed before any wave reads staged weights
  if (w == 0) __builtin_amdgcn_s_wait_tensorcnt(0);
  __syncthreads();

  for (int t = 0; t < T_STEPS; t++) {
    // stage x_t [16,3]
    if (threadIdx.x < 48) {
      int r = threadIdx.x / 3, k = threadIdx.x % 3;
      xs[r][k] = x[((long)t * BATCH + m0 + r) * 3 + k];
    }
    __syncthreads();

    // A fragments of h_{t-1} (shared across i,f,g,o)
    v16bf ha[4];
#pragma unroll
    for (int kt = 0; kt < 4; kt++) ha[kt] = load_a_frag(&hBF[0][0], HS, lane, kt * 32);

    // gates = x@W + h@U + b   (accumulator pre-loaded with bias + x@W; U from LDS)
    v8f acc[4];
#pragma unroll
    for (int X = 0; X < 4; X++) {
#pragma unroll
      for (int v = 0; v < 8; v++) {
        int r = rlo ? v : v + 8;
        acc[X][v] = bg[X] + xs[r][0] * Wg0[X] + xs[r][1] * Wg1[X] + xs[r][2] * Wg2[X];
      }
      int nt = X * 8 + w;
#pragma unroll
      for (int kt = 0; kt < 4; kt++)
        acc[X] = wmma_bf16(ha[kt], load_b_frag(U_lds, kt * 32 + nt, lane), acc[X]);
    }

    // pointwise LSTM update (c lives in registers in WMMA C/D layout)
#pragma unroll
    for (int v = 0; v < 8; v++) {
      float i_ = sigmf(acc[0][v]);
      float f_ = sigmf(acc[1][v]);
      float g_ = tanhfast(acc[2][v]);
      float o_ = sigmf(acc[3][v]);
      float c_ = f_ * creg[v] + i_ * g_;
      creg[v] = c_;
      hreg[v] = o_ * tanhfast(c_);
    }

    __syncthreads();  // all waves done reading old hBF (ha loaded) before overwrite
#pragma unroll
    for (int v = 0; v < 8; v++) {
      int r = rlo ? v : v + 8;
      long row = m0 + r;
      hBF[r][w * 16 + col16] = (__bf16)hreg[v];
      out[off_h + ((long)t * BATCH + row) * 128 + w * 16 + col16] = hreg[v];
      out[off_c + ((long)t * BATCH + row) * 128 + w * 16 + col16] = creg[v];
    }
    __syncthreads();  // new hBF visible

    // bottleneck bo = h_t @ Wb : wave w -> ntiles 2w, 2w+1 (Wb from L2)
    v16bf ha2[4];
#pragma unroll
    for (int kt = 0; kt < 4; kt++) ha2[kt] = load_a_frag(&hBF[0][0], HS, lane, kt * 32);
#pragma unroll
    for (int j = 0; j < 2; j++) {
      int nt = 2 * w + j;
      v8f bo = {0.f, 0.f, 0.f, 0.f, 0.f, 0.f, 0.f, 0.f};
#pragma unroll
      for (int kt = 0; kt < 4; kt++)
        bo = wmma_bf16(ha2[kt], load_b_frag(Wb_p, kt * 16 + nt, lane), bo);
#pragma unroll
      for (int v = 0; v < 8; v++) {
        int r = rlo ? v : v + 8;
        long row = m0 + r;
        boBF[r][nt * 16 + col16] = (__bf16)bo[v];
        out[off_bo + ((long)t * BATCH + row) * 256 + nt * 16 + col16] = bo[v];
      }
    }
    __syncthreads();  // boBF visible

    // A fragments of bo (shared by pc and hd)
    v16bf ba[8];
#pragma unroll
    for (int kt = 0; kt < 8; kt++) ba[kt] = load_a_frag(&boBF[0][0], 256, lane, kt * 32);

    // pc = bo @ Wpc + bpc : wave w -> ntiles 2w, 2w+1 (Wpc from LDS)
#pragma unroll
    for (int j = 0; j < 2; j++) {
      int nt = 2 * w + j;
      v8f pca;
#pragma unroll
      for (int v = 0; v < 8; v++) pca[v] = bpcv[j];
#pragma unroll
      for (int kt = 0; kt < 8; kt++)
        pca = wmma_bf16(ba[kt], load_b_frag(Wpc_lds, kt * 16 + nt, lane), pca);
#pragma unroll
      for (int v = 0; v < 8; v++) {
        int r = rlo ? v : v + 8;
        long row = m0 + r;
        out[off_pc + ((long)t * BATCH + row) * 256 + nt * 16 + col16] = pca[v];
      }
    }

    // hd = bo @ Whd + bhd : one padded 16-col ntile, wave 0 only (wave-uniform branch)
    if (w == 0) {
      v8f hda;
#pragma unroll
      for (int v = 0; v < 8; v++) hda[v] = bhdv;
#pragma unroll
      for (int kt = 0; kt < 8; kt++)
        hda = wmma_bf16(ba[kt], load_b_frag(Whd_p, kt, lane), hda);
      if (col16 < 12) {
#pragma unroll
        for (int v = 0; v < 8; v++) {
          int r = rlo ? v : v + 8;
          long row = m0 + r;
          out[((long)t * BATCH + row) * 12 + col16] = hda[v];
        }
      }
    }
    __syncthreads();  // boBF / xs reads done before next step overwrites
  }
}

extern "C" void kernel_launch(void* const* d_in, const int* in_sizes, int n_in,
                              void* d_out, int out_size, void* d_ws, size_t ws_size,
                              hipStream_t stream) {
  const float* x    = (const float*)d_in[0];
  const float* init = (const float*)d_in[1];
  const float* W    = (const float*)d_in[2];
  const float* U    = (const float*)d_in[3];
  const float* b    = (const float*)d_in[4];
  const float* Wh   = (const float*)d_in[5];
  const float* bh   = (const float*)d_in[6];
  const float* Wc   = (const float*)d_in[7];
  const float* bc   = (const float*)d_in[8];
  const float* Wb   = (const float*)d_in[9];
  const float* Wpc  = (const float*)d_in[10];
  const float* bpc  = (const float*)d_in[11];
  const float* Whd  = (const float*)d_in[12];
  const float* bhd  = (const float*)d_in[13];
  __bf16* ws = (__bf16*)d_ws;
  float* out = (float*)d_out;

  // one-time bf16 fragment-order weight packing (deterministic, runs every launch)
  int pack_threads = 256;
  int pack_blocks = (int)((WS_BF16_TOTAL + pack_threads - 1) / pack_threads);
  pack_weights<<<pack_blocks, pack_threads, 0, stream>>>(U, Wb, Wpc, Whd, Wh, Wc, init, ws);

  // persistent LSTM: 128 blocks x 8 waves, 256KB dynamic LDS for TDM-staged U/Wpc
  lstm_main<<<BATCH / 16, 256, LDS_DYN_BYTES, stream>>>(x, W, b, bh, bc, bpc, bhd, ws, out);
}
